// RetrievalDatabase_Duet_24876450579154
// MI455X (gfx1250) — matrix operand order repaired and checked by reading, not compiled, not run on hardware
//
#include <hip/hip_runtime.h>

// ---------------------------------------------------------------------------
// RetrievalDatabase_Duet — MI455X (gfx1250) implementation
//   prep : normalize queries -> bf16 table in ws
//   main : streamed bf16 WMMA GEMM over db tiles + fused norm + kinematic +
//          per-block top-4
//   merge: global top-4 reduction
// ---------------------------------------------------------------------------

#define N_DB   262144
#define BQ     64
#define DDIM   512
#define TILE_N 128
#define KC     32
#define ROWSTRIDE 20            // dwords per LDS tile row: 16 data + 4 pad (80B, 16B aligned)
#define NEG_BIG (-3.402823466e38f)

typedef __attribute__((ext_vector_type(16))) __bf16        v16bf;
typedef __attribute__((ext_vector_type(8)))  float         v8f;
typedef __attribute__((ext_vector_type(4)))  float         v4f;
typedef __attribute__((ext_vector_type(4)))  unsigned int  v4u;

union Frag { v4u h[2]; v16bf v; };

// Async global->LDS copy, 16B per lane (CDNA5 VGLOBAL async, GV mode).
// VDST = VGPR with LDS byte address, VADDR = 64-bit global address.
// Generic->LDS address: the LDS aperture truncates to addr[31:0] (ISA 10.2).
__device__ __forceinline__ void async_copy_b128(const void* gsrc, void* ldst) {
    unsigned ldsAddr = (unsigned)(size_t)ldst;
    asm volatile("global_load_async_to_lds_b128 %0, %1, off"
                 :: "v"(ldsAddr), "v"((unsigned long long)(size_t)gsrc)
                 : "memory");
}

__device__ __forceinline__ void wait_async0() {
#if __has_builtin(__builtin_amdgcn_s_wait_asynccnt)
    __builtin_amdgcn_s_wait_asynccnt(0);
#else
    asm volatile("s_wait_asynccnt 0" ::: "memory");
#endif
}

// Pack two fp32 into a bf16x2 dword using the VOP3P mixed-precision
// converters (V_FMA_MIXLO_BF16 / V_FMA_MIXHI_BF16): bf16(x*1+0) per half.
// Default OPSEL -> sources read as f32; RNE per MODE. 2 VALU per dword.
__device__ __forceinline__ unsigned int pack_bf16(float a, float b) {
    unsigned int d;
    asm("v_fma_mixlo_bf16 %0, %1, 1.0, 0\n\t"
        "v_fma_mixhi_bf16 %0, %2, 1.0, 0"
        : "=v"(d)
        : "v"(a), "v"(b));
    return d;
}

__device__ __forceinline__ void top4_insert(float v, int i, float* tv, int* ti) {
    if (v > tv[3]) {
        int p = 3;
        while (p > 0 && v > tv[p - 1]) { tv[p] = tv[p - 1]; ti[p] = ti[p - 1]; --p; }
        tv[p] = v; ti[p] = i;
    }
}

// ---------------------------------------------------------------------------
// Kernel 1: per-(space,query) normalization -> bf16 table [4][64][512]
// ---------------------------------------------------------------------------
__global__ __launch_bounds__(256)
void duet_prep(const float* __restrict__ q0, const float* __restrict__ q1,
               const float* __restrict__ q2, const float* __restrict__ q3,
               unsigned int* __restrict__ qn) {
    __shared__ float red[256];
    const int t = threadIdx.x;
    const int s = blockIdx.x >> 6;
    const int b = blockIdx.x & 63;
    const float* const qs[4] = {q0, q1, q2, q3};
    const float* q = qs[s] + b * DDIM;

    float a0 = q[t], a1 = q[t + 256];
    red[t] = a0 * a0 + a1 * a1;
    __syncthreads();
    for (int off = 128; off > 0; off >>= 1) {
        if (t < off) red[t] += red[t + off];
        __syncthreads();
    }
    const float scale = 1.0f / fmaxf(sqrtf(red[0]), 1e-8f);
    const float e0 = q[2 * t] * scale;
    const float e1 = q[2 * t + 1] * scale;
    qn[s * (BQ * DDIM / 2) + b * (DDIM / 2) + t] = pack_bf16(e0, e1);
}

// ---------------------------------------------------------------------------
// Kernel 2: main streamed GEMM + fusion + per-block top-4
// ---------------------------------------------------------------------------
__global__ __launch_bounds__(256)
void duet_main(const float* __restrict__ db0, const float* __restrict__ db1,
               const float* __restrict__ db2, const float* __restrict__ db3,
               const unsigned int* __restrict__ qn,
               const int* __restrict__ db_len, const int* __restrict__ q_len,
               float* __restrict__ candVals, int* __restrict__ candIdx) {
    __shared__ unsigned int dbTile[TILE_N * ROWSTRIDE];   // bf16 db chunk
    __shared__ unsigned int qTile[BQ * ROWSTRIDE];        // bf16 q chunk
    __shared__ float rowSumSq[TILE_N];
    __shared__ float comb[BQ * (TILE_N + 4)];
    __shared__ int   LqSh[BQ];

    const int t    = threadIdx.x;
    const int wv   = t >> 5;        // wave 0..7 -> owns N-subtile wv
    const int lane = t & 31;
    const int r    = lane & 15;
    const int hf   = lane >> 4;     // lane half (WMMA fragment layout)
    const int n0   = blockIdx.x * TILE_N;

    if (t < BQ) LqSh[t] = q_len[t];

    const float* const dbs[4] = {db0, db1, db2, db3};

    v8f zacc;
#pragma unroll
    for (int j = 0; j < 8; ++j) zacc[j] = 0.0f;

    v8f accD[4];
#pragma unroll
    for (int m = 0; m < 4; ++m) accD[m] = zacc;

    // Fully unrolled so dbs[s] folds to the kernel-arg pointer -> clang keeps
    // the db stream in the GLOBAL address space (global_load_b128, not flat).
#pragma unroll
    for (int s = 0; s < 4; ++s) {
        const float* __restrict__ db = dbs[s];
        __syncthreads();
        if (t < TILE_N) rowSumSq[t] = 0.0f;
        __syncthreads();

        v8f c[4];
#pragma unroll
        for (int m = 0; m < 4; ++m) c[m] = zacc;

        for (int kc = 0; kc < DDIM; kc += KC) {
            // ---- stage q chunk: bf16 table -> LDS via async DMA path ----
            {
                const int b  = t >> 2;
                const int ch = t & 3;
                const unsigned int* src =
                    qn + s * (BQ * DDIM / 2) + b * (DDIM / 2) + kc / 2 + ch * 4;
                unsigned int* dst = &qTile[b * ROWSTRIDE + ch * 4];
                async_copy_b128(src, dst);
            }
            // ---- stage db: fp32 global -> (sumsq, bf16 LDS) ----
            // Phase 1: issue all 4 loads (keeps 4 in flight), prefetch next chunk.
            v4f f[4];
#pragma unroll
            for (int p = 0; p < 4; ++p) {
                const int row = p * 32 + (t >> 3);   // 0..127
                const int cg  = t & 7;               // 8 float4 per row
                const float* src = db + (size_t)(n0 + row) * DDIM + kc + cg * 4;
                if (cg == 0 && kc + KC < DDIM) __builtin_prefetch(src + KC, 0, 3);
                f[p] = *(const v4f*)src;
            }
            // Phase 2: sumsq accumulate + bf16 pack + LDS store.
#pragma unroll
            for (int p = 0; p < 4; ++p) {
                const int row = p * 32 + (t >> 3);
                const int cg  = t & 7;
                atomicAdd(&rowSumSq[row],
                          f[p][0] * f[p][0] + f[p][1] * f[p][1] +
                          f[p][2] * f[p][2] + f[p][3] * f[p][3]);
                dbTile[row * ROWSTRIDE + cg * 2 + 0] = pack_bf16(f[p][0], f[p][1]);
                dbTile[row * ROWSTRIDE + cg * 2 + 1] = pack_bf16(f[p][2], f[p][3]);
            }
            wait_async0();
            __syncthreads();

            // ---- WMMA: C[m][wv] += A[m] * B[wv] ----
            // B fragment (32x16 bf16): lane r = column, lanes0-15 K{0..7,16..23}, lanes16-31 K{8..15,24..31}
            const unsigned int* bp = &dbTile[(16 * wv + r) * ROWSTRIDE + hf * 4];
            Frag fb;
            fb.h[0] = *(const v4u*)bp;
            fb.h[1] = *(const v4u*)(bp + 8);
#pragma unroll
            for (int m = 0; m < 4; ++m) {
                // A fragment (16x32 bf16): lanes0-15 K{0..7,16..23}, lanes16-31 K{8..15,24..31}
                const unsigned int* ap = &qTile[(16 * m + r) * ROWSTRIDE];
                Frag fa;
                fa.h[0] = *(const v4u*)(ap + hf * 4);
                fa.h[1] = *(const v4u*)(ap + 8 + hf * 4);
                c[m] = __builtin_amdgcn_wmma_f32_16x16x32_bf16(
                    false, fa.v, false, fb.v, (short)0, c[m], false, false);
            }
            __syncthreads();
        }

        // scale this space's partial dot by 1/max(||db_row||, eps); N index is lane-resident
        const float nrm   = sqrtf(rowSumSq[16 * wv + r]);
        const float scale = 1.0f / fmaxf(nrm, 1e-8f);
#pragma unroll
        for (int m = 0; m < 4; ++m)
#pragma unroll
            for (int j = 0; j < 8; ++j)
                accD[m][j] += c[m][j] * scale;
    }

    // ---- kinematic weighting + dump combined scores to LDS ----
    const int   n_local = 16 * wv + r;
    const float Ldb     = (float)db_len[n0 + n_local];
#pragma unroll
    for (int m = 0; m < 4; ++m)
#pragma unroll
        for (int j = 0; j < 8; ++j) {
            const int   b    = 16 * m + j + 8 * hf;   // C layout: VGPR j -> M=j (+8 for hi half)
            const float Lq   = (float)LqSh[b];
            const float dmax = fmaxf(fmaxf(Ldb, Lq), 1.0f);
            const float rel  = fabsf(Ldb - Lq) / dmax;
            const float kin  = __expf(-0.1f * rel);
            comb[b * (TILE_N + 4) + n_local] = accD[m][j] * 0.25f * kin;
        }
    __syncthreads();

    // ---- per-block top-4 per query ----
    if (t < BQ) {
        float tv[4] = {NEG_BIG, NEG_BIG, NEG_BIG, NEG_BIG};
        int   ti[4] = {0, 0, 0, 0};
        for (int col = 0; col < TILE_N; ++col)
            top4_insert(comb[t * (TILE_N + 4) + col], n0 + col, tv, ti);
        const size_t base = ((size_t)blockIdx.x * BQ + t) * 4;
#pragma unroll
        for (int j = 0; j < 4; ++j) {
            candVals[base + j] = tv[j];
            candIdx[base + j]  = ti[j];
        }
    }
}

// ---------------------------------------------------------------------------
// Kernel 3: reduce 2048 blocks x 4 candidates -> global top-4 per query
// ---------------------------------------------------------------------------
__global__ __launch_bounds__(256)
void duet_merge(const float* __restrict__ candVals, const int* __restrict__ candIdx,
                float* __restrict__ out) {
    __shared__ float sv[1024];
    __shared__ int   si[1024];
    const int b = blockIdx.x;
    const int t = threadIdx.x;
    const int nBlocks = N_DB / TILE_N;   // 2048

    float tv[4] = {NEG_BIG, NEG_BIG, NEG_BIG, NEG_BIG};
    int   ti[4] = {0, 0, 0, 0};
    for (int e = t; e < nBlocks * 4; e += 256) {
        const int blk = e >> 2, j = e & 3;
        const size_t off = ((size_t)blk * BQ + b) * 4 + j;
        top4_insert(candVals[off], candIdx[off], tv, ti);
    }
#pragma unroll
    for (int j = 0; j < 4; ++j) { sv[t * 4 + j] = tv[j]; si[t * 4 + j] = ti[j]; }
    __syncthreads();

    if (t == 0) {
        float fv[4] = {NEG_BIG, NEG_BIG, NEG_BIG, NEG_BIG};
        int   fi[4] = {0, 0, 0, 0};
        for (int e = 0; e < 1024; ++e) top4_insert(sv[e], si[e], fv, fi);
#pragma unroll
        for (int j = 0; j < 4; ++j) {
            out[b * 4 + j]               = fv[j];            // top_vals
            out[BQ * 4 + b * 4 + j]      = (float)fi[j];     // top_idx (tuple, flat)
        }
    }
}

// ---------------------------------------------------------------------------
extern "C" void kernel_launch(void* const* d_in, const int* in_sizes, int n_in,
                              void* d_out, int out_size, void* d_ws, size_t ws_size,
                              hipStream_t stream) {
    const float* db0 = (const float*)d_in[0];
    const float* db1 = (const float*)d_in[1];
    const float* db2 = (const float*)d_in[2];
    const float* db3 = (const float*)d_in[3];
    const float* q0  = (const float*)d_in[4];
    const float* q1  = (const float*)d_in[5];
    const float* q2  = (const float*)d_in[6];
    const float* q3  = (const float*)d_in[7];
    const int* db_len = (const int*)d_in[8];
    const int* q_len  = (const int*)d_in[9];

    char* ws = (char*)d_ws;
    unsigned int* qn = (unsigned int*)ws;                       // 256 KB bf16 query table
    float* candVals  = (float*)(ws + 262144);                   // 2 MB
    int*   candIdx   = (int*)(ws + 262144 + 2097152);           // 2 MB
    float* out       = (float*)d_out;

    duet_prep<<<4 * BQ, 256, 0, stream>>>(q0, q1, q2, q3, qn);
    duet_main<<<N_DB / TILE_N, 256, 0, stream>>>(db0, db1, db2, db3, qn,
                                                 db_len, q_len, candVals, candIdx);
    duet_merge<<<BQ, 256, 0, stream>>>(candVals, candIdx, out);
}
